// BranchLayer_50448685859461
// MI455X (gfx1250) — matrix operand order repaired and checked by reading, not compiled, not run on hardware
//
#include <hip/hip_runtime.h>
#include <hip/hip_bf16.h>

#define N_IN   4096
#define N_NPB  64
#define N_COLS 16384
#define BATCH  128

typedef __attribute__((ext_vector_type(16))) _Float16 v16h;
typedef __attribute__((ext_vector_type(8)))  float    v8f;

// ---------------------------------------------------------------------------
// Kernel 1: transpose x[128][4096] (f32) -> xT[4096][128] (f16)
// Makes x[:, i] contiguous (256B rows) so the random gather in the main
// kernel coalesces along the batch dimension, and halves gather bytes.
// ---------------------------------------------------------------------------
__global__ __launch_bounds__(256) void transpose_x_f16(
    const float* __restrict__ x, _Float16* __restrict__ xT) {
  __shared__ float tile[32][33];                  // +1 pad kills bank conflicts
  const int i0 = blockIdx.x * 32;                 // feature tile
  const int b0 = blockIdx.y * 32;                 // batch tile
  const int tx = threadIdx.x, ty = threadIdx.y;   // block = 32 x 8
#pragma unroll
  for (int r = 0; r < 32; r += 8)
    tile[ty + r][tx] = x[(size_t)(b0 + ty + r) * N_IN + i0 + tx];
  __syncthreads();
#pragma unroll
  for (int r = 0; r < 32; r += 8)
    xT[(size_t)(i0 + ty + r) * BATCH + b0 + tx] = (_Float16)tile[tx][ty + r];
}

// ---------------------------------------------------------------------------
// Kernel 2: gather + WMMA.
// Block (256 thr = 8 wave32) owns 16 output columns x all 128 batch rows;
// wave mt handles batch subtile [16*mt, 16*mt+16).
// K inflated to 1024 slots (16 cols x 64 k); K-chunk kc covers column
// c' = kc>>1, k-range (kc&1)*32..+32; B is block-diagonal (only output
// column n == c' nonzero) holding the last-write-wins-deduped f16 weights.
// Fragment layouts (wave32; D per CDNA5 ISA 7.12.2, A/B k order is this code's contiguous choice):
//   A f16 16x32 : lane L: m=L&15; elem e -> k = ((L>>4)<<4) + e
//   B f16 32x16 : lane L: n=L&15; elem e -> k = ((L>>4)<<4) + e
//   D f32 16x16 : lane L: n=L&15; vgpr r -> row = r + 8*(L>>4)
// idx_s holds PRE-SCALED byte offsets (i * BATCH * 2) to shave one VALU op
// off every gathered element; scaling is injective so the dedup compare in
// phase 2 is unaffected.
// ---------------------------------------------------------------------------
__global__ __launch_bounds__(256) void branch_gather_wmma(
    const float* __restrict__ w, const int* __restrict__ idx,
    const _Float16* __restrict__ xT, float* __restrict__ out) {
  __shared__ int      idx_s[16][N_NPB];   // 4 KB  idx_s[c'][k] = row byte offset
  __shared__ _Float16 we_s[16][N_NPB];    // 2 KB  deduped f16 weights

  const int cbase = blockIdx.x * 16;
  const int tid   = threadIdx.x;

  // ---- phase 1: stage idx tile as byte offsets (coalesced; single-use -> NT)
#pragma unroll
  for (int t = 0; t < 4; ++t) {
    const int e = tid + t * 256;          // e = k*16 + c'
    const int k = e >> 4, cp = e & 15;
    idx_s[cp][k] = __builtin_nontemporal_load(&idx[k * N_COLS + cbase + cp])
                   << 8;                  // i * BATCH * sizeof(_Float16)
  }
  __syncthreads();

  // ---- phase 2: last-write-wins mask + f16 weights (torch index_put semantics)
#pragma unroll 1
  for (int t = 0; t < 4; ++t) {
    const int e = tid + t * 256;
    const int k = e >> 4, cp = e & 15;
    const int iv = idx_s[cp][k];
    bool keep = true;
    for (int k2 = k + 1; k2 < N_NPB; ++k2) keep &= (idx_s[cp][k2] != iv);
    const float wv = __builtin_nontemporal_load(&w[k * N_COLS + cbase + cp]);
    we_s[cp][k] = keep ? (_Float16)wv : (_Float16)0.f;
  }
  __syncthreads();

  // ---- phase 3: gather + WMMA accumulate over 32 K-chunks
  const int lane = tid & 31;
  const int mt   = tid >> 5;        // wave id = batch subtile
  const int m    = lane & 15;
  const int hh   = lane >> 4;       // lane half (0/1)
  const int b    = mt * 16 + m;     // batch row this lane owns in A

  const char* xTb = (const char*)xT + b * sizeof(_Float16);

  v8f acc = {};
#pragma unroll 4
  for (int kc = 0; kc < 32; ++kc) {
    const int cp    = kc >> 1;
    const int kbase = (kc & 1) * 32;

    // A fragment: pure gather (16 lanes x 2B contiguous per half-wave)
    v16h a;
#pragma unroll
    for (int e = 0; e < 16; ++e) {
      const int klocal = (hh << 4) + e;
      const int off = idx_s[cp][kbase + klocal];        // LDS broadcast read
      a[e] = *(const _Float16*)(xTb + off);
    }

    // B fragment: block-diagonal weights, nonzero only for column n == cp.
    // Consecutive e -> two ds_load_b128; lane select keeps EXEC intact.
    v16h bf;
#pragma unroll
    for (int e = 0; e < 16; ++e) {
      const _Float16 wv = we_s[cp][kbase + hh * 16 + e];
      bf[e] = (m == cp) ? wv : (_Float16)0.f;
    }

    acc = __builtin_amdgcn_wmma_f32_16x16x32_f16(
        /*neg_a=*/false, a, /*neg_b=*/false, bf,
        /*c_mod=*/(short)0, acc, /*reuse_a=*/false, /*reuse_b=*/false);
  }

  // ---- store D (write-once 8MB -> non-temporal, keep L2 for the xT gather)
#pragma unroll
  for (int r = 0; r < 8; ++r) {
    __builtin_nontemporal_store(
        acc[r], &out[(size_t)(mt * 16 + r + 8 * hh) * N_COLS + cbase + m]);
  }
}

// ---------------------------------------------------------------------------
extern "C" void kernel_launch(void* const* d_in, const int* in_sizes, int n_in,
                              void* d_out, int out_size, void* d_ws, size_t ws_size,
                              hipStream_t stream) {
  const float* x   = (const float*)d_in[0];   // [128, 4096] f32
  const float* w   = (const float*)d_in[1];   // [64, 16384] f32
  const int*   idx = (const int*)d_in[2];     // [64, 16384] i32
  float*       out = (float*)d_out;           // [128, 16384] f32

  _Float16* xT = (_Float16*)d_ws;             // 4096*128*2 = 1 MB scratch

  transpose_x_f16<<<dim3(N_IN / 32, BATCH / 32), dim3(32, 8), 0, stream>>>(x, xT);
  branch_gather_wmma<<<dim3(N_COLS / 16), dim3(256), 0, stream>>>(w, idx, xT, out);
}
